// TransformerBlock_9088150799005
// MI455X (gfx1250) — matrix operand order repaired and checked by reading, not compiled, not run on hardware
//
#include <hip/hip_runtime.h>
#include <math.h>

// ---------------------------------------------------------------------------
// CDNA5 (gfx1250) wave32 WMMA transformer block, bf16 datapath,
// double-buffered async global->LDS tile staging (ASYNCcnt pipelining).
// ---------------------------------------------------------------------------

typedef __attribute__((ext_vector_type(16))) __bf16 v16bf;
typedef __attribute__((ext_vector_type(8)))  float  v8f;

union BfPack { v16bf v; unsigned int u[8]; };
union BfPair { __bf16 b[2]; unsigned int u; };

typedef __attribute__((address_space(3))) char lds_char;

__device__ __forceinline__ unsigned ldsaddr(const void* p) {
  return (unsigned)(unsigned long long)(const lds_char*)p;
}

// Per-lane async DMA: 16 bytes global(SADDR base + 32-bit voff) -> LDS offset.
__device__ __forceinline__ void async_load_b128(unsigned lds_off,
                                                const void* base,
                                                unsigned voff) {
  asm volatile("global_load_async_to_lds_b128 %0, %1, %2"
               :: "v"(lds_off), "v"(voff), "s"(base)
               : "memory");
}

__device__ __forceinline__ void wait_async0() {
  asm volatile("s_wait_asynccnt 0" ::: "memory");
}

__device__ __forceinline__ v8f wmma_bf16(v16bf a, v16bf b, v8f c) {
  return __builtin_amdgcn_wmma_f32_16x16x32_bf16(
      false, a, false, b, (short)0, c, false, false);
}

__device__ __forceinline__ int afrag_k(int v, int kh) {
  return 2 * v + 16 * kh;
}

// ---------------------------------------------------------------------------
// One-time fp32 -> bf16 conversion.
// ---------------------------------------------------------------------------
__global__ __launch_bounds__(256) void cvt_bf16_kernel(
    const float* __restrict__ in, __bf16* __restrict__ out, int n4)
{
  int i = blockIdx.x * 256 + threadIdx.x;
  if (i < n4) {
    float4 f = ((const float4*)in)[i];
    union { __bf16 b[4]; uint2 u; } p;
    p.b[0] = (__bf16)f.x; p.b[1] = (__bf16)f.y;
    p.b[2] = (__bf16)f.z; p.b[3] = (__bf16)f.w;
    ((uint2*)out)[i] = p.u;
  }
}

// ---------------------------------------------------------------------------
// One-time W[K][N] fp32 -> Wt[N][K] bf16 (tiled transpose).
// ---------------------------------------------------------------------------
__global__ __launch_bounds__(256) void transpose_bf16_kernel(
    const float* __restrict__ W, __bf16* __restrict__ Wt, int K, int N)
{
  __shared__ float t[32][33];
  const int n0 = blockIdx.x * 32;
  const int k0 = blockIdx.y * 32;
  const int tx = threadIdx.x & 31;
  const int ty = threadIdx.x >> 5;
#pragma unroll
  for (int i = 0; i < 4; ++i) {
    int k = ty + i * 8;
    t[k][tx] = W[(size_t)(k0 + k) * N + (n0 + tx)];
  }
  __syncthreads();
#pragma unroll
  for (int i = 0; i < 4; ++i) {
    int n = ty + i * 8;
    Wt[(size_t)(n0 + n) * K + (k0 + tx)] = (__bf16)t[tx][n];
  }
}

// ---------------------------------------------------------------------------
// GEMM: C[M,N] = act(A[M,K] @ Wt[N,K]^T + bias[N]).  Compile-time epilogue.
// Double-buffered async tile staging; block 256 thr (8 waves), 128x128 tile.
// ---------------------------------------------------------------------------
#define BM 128
#define BN 128
#define BK 32
#define LDST 48   // LDS k-stride (elements): 96 B, multiple of 16 B

template <int WF, int WB, int RELU>
__global__ __launch_bounds__(256) void gemm_bias_kernel(
    const __bf16* __restrict__ A, const __bf16* __restrict__ Wt,
    const float* __restrict__ bias,
    float* __restrict__ Cf, __bf16* __restrict__ Cb,
    int N, int K)
{
  __shared__ __bf16 sA[2][BM * LDST];
  __shared__ __bf16 sB[2][BN * LDST];

  const int tid  = threadIdx.x;
  const int lane = tid & 31;
  const int wave = tid >> 5;
  const int wrow = wave >> 1;
  const int wcol = wave & 1;
  const int lm   = lane & 15;
  const int kh   = lane >> 4;
  const int bm0  = blockIdx.y * BM;
  const int bn0  = blockIdx.x * BN;

  const int sr = tid >> 2;       // staging row (pass 0), +64 for pass 1
  const int sq = tid & 3;        // 16-byte quarter

  auto stage = [&](int buf, int k0) {
#pragma unroll
    for (int i = 0; i < 2; ++i) {
      int r = sr + i * 64;
      async_load_b128(ldsaddr(&sA[buf][r * LDST + sq * 8]), A,
                      (unsigned)(((size_t)(bm0 + r) * K + k0 + sq * 8) * 2));
      async_load_b128(ldsaddr(&sB[buf][r * LDST + sq * 8]), Wt,
                      (unsigned)(((size_t)(bn0 + r) * K + k0 + sq * 8) * 2));
    }
  };

  v8f acc[2][4];
#pragma unroll
  for (int i = 0; i < 2; ++i)
#pragma unroll
    for (int j = 0; j < 4; ++j)
#pragma unroll
      for (int r = 0; r < 8; ++r) acc[i][j][r] = 0.0f;

  stage(0, 0);
  const int nsteps = K / BK;
  for (int s = 0; s < nsteps; ++s) {
    wait_async0();          // tile s resident in LDS (this wave's DMAs done)
    __syncthreads();        // ...and every wave's; prev reads of other buf done
    if (s + 1 < nsteps) stage((s + 1) & 1, (s + 1) * BK);  // overlap with WMMA

    const __bf16* pa = sA[s & 1];
    const __bf16* pb = sB[s & 1];

    v16bf afrag[2], bfrag[4];
#pragma unroll
    for (int mf = 0; mf < 2; ++mf) {
      const int row = wrow * 32 + mf * 16 + lm;
      BfPack p;
#pragma unroll
      for (int v = 0; v < 8; ++v)
        p.u[v] = *(const unsigned int*)&pa[row * LDST + afrag_k(v, kh)];
      afrag[mf] = p.v;
    }
#pragma unroll
    for (int nf = 0; nf < 4; ++nf) {
      const int col = wcol * 64 + nf * 16 + lm;
      BfPack p;
#pragma unroll
      for (int v = 0; v < 8; ++v)
        p.u[v] = *(const unsigned int*)&pb[col * LDST + (2 * v + 16 * kh)];
      bfrag[nf] = p.v;
    }
#pragma unroll
    for (int mf = 0; mf < 2; ++mf)
#pragma unroll
      for (int nf = 0; nf < 4; ++nf)
        acc[mf][nf] = wmma_bf16(afrag[mf], bfrag[nf], acc[mf][nf]);
  }

  // Branch-free epilogue: hoisted base pointers, row stride N.
#pragma unroll
  for (int mf = 0; mf < 2; ++mf)
#pragma unroll
    for (int nf = 0; nf < 4; ++nf) {
      const int row0 = bm0 + wrow * 32 + mf * 16 + 8 * kh;
      const int col  = bn0 + wcol * 64 + nf * 16 + lm;
      const float bc = bias[col];
      float*  cf = Cf + (size_t)row0 * N + col;
      __bf16* cb = Cb + (size_t)row0 * N + col;
#pragma unroll
      for (int r = 0; r < 8; ++r) {
        float v = acc[mf][nf][r] + bc;
        if (RELU) v = fmaxf(v, 0.0f);
        if (WF) cf[(size_t)r * N] = v;
        if (WB) cb[(size_t)r * N] = (__bf16)v;
      }
    }
}

// ---------------------------------------------------------------------------
// Flash attention on bf16 Q/K/V. block = (b*h, 128 q rows), 8 waves x 16 rows.
// Double-buffered KV chunks: K async-DMA'd, V staged transposed.
// ---------------------------------------------------------------------------
#define ATT_H  12
#define ATT_DK 64
#define ATT_D  768
#define LKK 64   // sK  [kv][dk] stride (128 B rows, 16B-aligned for async)
#define LVT 40   // sVt [dk][kv] stride
#define LPP 40   // sP per-wave [q][kv] stride

__global__ __launch_bounds__(256) void attention_kernel(
    const __bf16* __restrict__ Q, const __bf16* __restrict__ Kb,
    const __bf16* __restrict__ Vb, __bf16* __restrict__ Ob, int S)
{
  __shared__ __bf16 sK[2][32 * LKK];
  __shared__ __bf16 sVt[2][ATT_DK * LVT];
  __shared__ __bf16 sP[8 * 16 * LPP];

  const int bh    = blockIdx.x;
  const int b     = bh / ATT_H;
  const int h     = bh % ATT_H;
  const int qbase = blockIdx.y * 128;
  const int tid   = threadIdx.x;
  const int lane  = tid & 31;
  const int wave  = tid >> 5;
  const int lm    = lane & 15;
  const int kh    = lane >> 4;

  const __bf16* Qp = Q  + (size_t)b * S * ATT_D + h * ATT_DK;
  const __bf16* Kp = Kb + (size_t)b * S * ATT_D + h * ATT_DK;
  const __bf16* Vp = Vb + (size_t)b * S * ATT_D + h * ATT_DK;

  const int rk = tid >> 3;   // K staging: kv row
  const int qk = tid & 7;    // K staging: 16-byte eighth of the row

  auto stage = [&](int buf, int kc) {
    // K chunk (32x64 bf16 = 256 x 16B): one async b128 per thread.
    async_load_b128(ldsaddr(&sK[buf][rk * LKK + qk * 8]), Kp,
                    (unsigned)(((size_t)(kc + rk) * ATT_D + qk * 8) * 2));
    // V chunk staged transposed [dk][kv] (shuffled, so synchronous path).
#pragma unroll
    for (int i = 0; i < 2; ++i) {
      int idx = tid + i * 256;
      int r   = idx >> 4;            // kv 0..31
      int c   = (idx & 15) * 4;      // dk 0..60
      uint2 uv = *(const uint2*)(Vp + (size_t)(kc + r) * ATT_D + c);
      BfPair t0; t0.u = uv.x;
      BfPair t1; t1.u = uv.y;
      __bf16* dv = &sVt[buf][0];
      dv[(c + 0) * LVT + r] = t0.b[0];
      dv[(c + 1) * LVT + r] = t0.b[1];
      dv[(c + 2) * LVT + r] = t1.b[0];
      dv[(c + 3) * LVT + r] = t1.b[1];
    }
  };

  // Q fragments (A layout) straight from bf16 global (pairs 4B-aligned).
  v16bf qf[2];
  {
    const int qrow = qbase + wave * 16 + lm;
#pragma unroll
    for (int hf = 0; hf < 2; ++hf) {
      BfPack p;
#pragma unroll
      for (int v = 0; v < 8; ++v) {
        int kk = afrag_k(v, kh) + hf * 32;
        p.u[v] = *(const unsigned int*)(Qp + (size_t)qrow * ATT_D + kk);
      }
      qf[hf] = p.v;
    }
  }

  float m[8], l[8];
  v8f o[4];
#pragma unroll
  for (int r = 0; r < 8; ++r) { m[r] = -1e30f; l[r] = 0.0f; }
#pragma unroll
  for (int nf = 0; nf < 4; ++nf)
#pragma unroll
    for (int r = 0; r < 8; ++r) o[nf][r] = 0.0f;

  stage(0, 0);
  const int nC = S / 32;
  for (int c = 0; c < nC; ++c) {
    wait_async0();
    __syncthreads();
    if (c + 1 < nC) stage((c + 1) & 1, (c + 1) * 32);   // overlap with compute

    const __bf16* pk = &sK[c & 1][0];
    const __bf16* pv = &sVt[c & 1][0];

    // scores 16x32 = Qf(16x64) @ K_chunk^T
    float sc[2][8];
#pragma unroll
    for (int t = 0; t < 2; ++t) {
      v8f s;
#pragma unroll
      for (int r = 0; r < 8; ++r) s[r] = 0.0f;
#pragma unroll
      for (int hf = 0; hf < 2; ++hf) {
        BfPack p;
#pragma unroll
        for (int v = 0; v < 8; ++v)
          p.u[v] = *(const unsigned int*)&pk[(t * 16 + lm) * LKK + hf * 32 + (2 * v + 16 * kh)];
        s = wmma_bf16(qf[hf], p.v, s);
      }
#pragma unroll
      for (int r = 0; r < 8; ++r) sc[t][r] = s[r] * 0.125f;   // 1/sqrt(64)
    }

    // online softmax; row r+8*kh lives within one 16-lane half
#pragma unroll
    for (int r = 0; r < 8; ++r) {
      float mx = fmaxf(sc[0][r], sc[1][r]);
#pragma unroll
      for (int d = 8; d >= 1; d >>= 1) mx = fmaxf(mx, __shfl_xor(mx, d, 32));
      float mn = fmaxf(m[r], mx);
      float scale = __expf(m[r] - mn);
      sc[0][r] = __expf(sc[0][r] - mn);
      sc[1][r] = __expf(sc[1][r] - mn);
      float rs = sc[0][r] + sc[1][r];
#pragma unroll
      for (int d = 8; d >= 1; d >>= 1) rs += __shfl_xor(rs, d, 32);
      l[r] = l[r] * scale + rs;
      m[r] = mn;
#pragma unroll
      for (int nf = 0; nf < 4; ++nf) o[nf][r] *= scale;
    }

    // P: C layout -> wave-private LDS -> A layout
    __bf16* pw = &sP[wave * 16 * LPP];
#pragma unroll
    for (int t = 0; t < 2; ++t)
#pragma unroll
      for (int r = 0; r < 8; ++r)
        pw[(r + 8 * kh) * LPP + t * 16 + lm] = (__bf16)sc[t][r];

    v16bf pf;
    {
      BfPack p;
#pragma unroll
      for (int v = 0; v < 8; ++v)
        p.u[v] = *(const unsigned int*)&pw[lm * LPP + afrag_k(v, kh)];
      pf = p.v;
    }

    // O += P(16x32) @ V(32x64)
#pragma unroll
    for (int nf = 0; nf < 4; ++nf) {
      BfPack p;
#pragma unroll
      for (int v = 0; v < 8; ++v)
        p.u[v] = *(const unsigned int*)&pv[(nf * 16 + lm) * LVT + (2 * v + 16 * kh)];
      o[nf] = wmma_bf16(pf, p.v, o[nf]);
    }
  }

  // Epilogue: hoisted base pointer, row stride D.
  {
    const int row0 = qbase + wave * 16 + 8 * kh;
    __bf16* ob = Ob + ((size_t)b * S + row0) * ATT_D + h * ATT_DK + lm;
#pragma unroll
    for (int nf = 0; nf < 4; ++nf) {
      float invl;
#pragma unroll
      for (int r = 0; r < 8; ++r) {
        invl = 1.0f / l[r];
        ob[(size_t)r * ATT_D + nf * 16] = (__bf16)(o[nf][r] * invl);
      }
    }
  }
}

// ---------------------------------------------------------------------------
// out = LayerNorm(x + y) * g + be   (fp32; optional bf16 mirror, templated)
// ---------------------------------------------------------------------------
template <int WB>
__global__ __launch_bounds__(256) void add_layernorm_kernel(
    const float* __restrict__ x, const float* __restrict__ y,
    const float* __restrict__ g, const float* __restrict__ be,
    float* __restrict__ out, __bf16* __restrict__ outb, int D)
{
  __shared__ float ssum[8];
  __shared__ float ssq[8];
  const int tid  = threadIdx.x;
  const int lane = tid & 31;
  const int wave = tid >> 5;
  const size_t row = blockIdx.x;
  const float* xr = x + row * D;
  const float* yr = y + row * D;

  float vals[3];
  float sum = 0.0f, sq = 0.0f;
#pragma unroll
  for (int i = 0; i < 3; ++i) {
    int c = tid + i * 256;
    float v = xr[c] + yr[c];
    vals[i] = v; sum += v; sq += v * v;
  }
#pragma unroll
  for (int d = 16; d >= 1; d >>= 1) {
    sum += __shfl_xor(sum, d, 32);
    sq  += __shfl_xor(sq,  d, 32);
  }
  if (lane == 0) { ssum[wave] = sum; ssq[wave] = sq; }
  __syncthreads();
  float ts = 0.0f, tq = 0.0f;
#pragma unroll
  for (int w = 0; w < 8; ++w) { ts += ssum[w]; tq += ssq[w]; }
  const float invD = 1.0f / (float)D;
  float mu  = ts * invD;
  float var = tq * invD - mu * mu;
  float rstd = rsqrtf(var + 1e-5f);
#pragma unroll
  for (int i = 0; i < 3; ++i) {
    int c = tid + i * 256;
    float v = (vals[i] - mu) * rstd * g[c] + be[c];
    out[row * D + c] = v;
    if (WB) outb[row * D + c] = (__bf16)v;
  }
}

// ---------------------------------------------------------------------------
// Launch: cvt/transpose -> QKV -> attention -> Wo -> LN1 -> FFN -> LN2
// ---------------------------------------------------------------------------
extern "C" void kernel_launch(void* const* d_in, const int* in_sizes, int n_in,
                              void* d_out, int out_size, void* d_ws, size_t ws_size,
                              hipStream_t stream) {
  (void)in_sizes; (void)n_in; (void)out_size; (void)ws_size;

  const float* x   = (const float*)d_in[0];
  const float* Wq  = (const float*)d_in[1];  const float* bq = (const float*)d_in[2];
  const float* Wk  = (const float*)d_in[3];  const float* bk = (const float*)d_in[4];
  const float* Wv  = (const float*)d_in[5];  const float* bv = (const float*)d_in[6];
  const float* Wo  = (const float*)d_in[7];  const float* bo = (const float*)d_in[8];
  const float* W1  = (const float*)d_in[9];  const float* b1 = (const float*)d_in[10];
  const float* W2  = (const float*)d_in[11]; const float* b2 = (const float*)d_in[12];
  const float* g1  = (const float*)d_in[13]; const float* be1 = (const float*)d_in[14];
  const float* g2  = (const float*)d_in[15]; const float* be2 = (const float*)d_in[16];

  const int Bb = 4, S = 2048, D = 768, FF = 3072;
  const size_t MR = (size_t)Bb * S;          // 8192 rows

  // ---- workspace layout ----
  __bf16* R0  = (__bf16*)d_ws;               // MR*FF elems: Q|K|V|At, later ff1
  __bf16* Qb  = R0;
  __bf16* Kb  = Qb + MR * D;
  __bf16* Vb  = Kb + MR * D;
  __bf16* At  = Vb + MR * D;
  __bf16* ff1 = R0;
  __bf16* xbf = R0  + MR * FF;
  __bf16* Wqt = xbf + MR * D;
  __bf16* Wkt = Wqt + (size_t)D * D;
  __bf16* Wvt = Wkt + (size_t)D * D;
  __bf16* Wot = Wvt + (size_t)D * D;
  __bf16* W1t = Wot + (size_t)D * D;
  __bf16* W2t = W1t + (size_t)D * FF;
  __bf16* Hbf = W2t + (size_t)D * FF;
  float*  Ao  = (float*)(Hbf + MR * D);
  float*  Hb  = Ao + MR * D;
  float*  F2  = Ao;

  dim3 blk(256);

  {
    int n4 = (int)(MR * D / 4);
    cvt_bf16_kernel<<<dim3((n4 + 255) / 256), blk, 0, stream>>>(x, xbf, n4);
  }
  transpose_bf16_kernel<<<dim3(D / 32, D / 32),  blk, 0, stream>>>(Wq, Wqt, D, D);
  transpose_bf16_kernel<<<dim3(D / 32, D / 32),  blk, 0, stream>>>(Wk, Wkt, D, D);
  transpose_bf16_kernel<<<dim3(D / 32, D / 32),  blk, 0, stream>>>(Wv, Wvt, D, D);
  transpose_bf16_kernel<<<dim3(D / 32, D / 32),  blk, 0, stream>>>(Wo, Wot, D, D);
  transpose_bf16_kernel<<<dim3(FF / 32, D / 32), blk, 0, stream>>>(W1, W1t, D, FF);
  transpose_bf16_kernel<<<dim3(D / 32, FF / 32), blk, 0, stream>>>(W2, W2t, FF, D);

  // QKV projections (bf16 out)
  gemm_bias_kernel<0,1,0><<<dim3(D / BN, MR / BM), blk, 0, stream>>>(xbf, Wqt, bq, nullptr, Qb, D, D);
  gemm_bias_kernel<0,1,0><<<dim3(D / BN, MR / BM), blk, 0, stream>>>(xbf, Wkt, bk, nullptr, Kb, D, D);
  gemm_bias_kernel<0,1,0><<<dim3(D / BN, MR / BM), blk, 0, stream>>>(xbf, Wvt, bv, nullptr, Vb, D, D);

  // flash attention
  attention_kernel<<<dim3(Bb * ATT_H, S / 128), blk, 0, stream>>>(Qb, Kb, Vb, At, S);

  // output projection (fp32) + residual LN (fp32 + bf16 mirror)
  gemm_bias_kernel<1,0,0><<<dim3(D / BN, MR / BM), blk, 0, stream>>>(At, Wot, bo, Ao, nullptr, D, D);
  add_layernorm_kernel<1><<<dim3((unsigned)MR), blk, 0, stream>>>(x, Ao, g1, be1, Hb, Hbf, D);

  // FFN
  gemm_bias_kernel<0,1,1><<<dim3(FF / BN, MR / BM), blk, 0, stream>>>(Hbf, W1t, b1, nullptr, ff1, FF, D);
  gemm_bias_kernel<1,0,0><<<dim3(D / BN, MR / BM), blk, 0, stream>>>(ff1, W2t, b2, F2, nullptr, D, FF);
  add_layernorm_kernel<0><<<dim3((unsigned)MR), blk, 0, stream>>>(Hb, F2, g2, be2, (float*)d_out, nullptr, D);
}
